// RWKVv7Separator_78280073937594
// MI455X (gfx1250) — matrix operand order repaired and checked by reading, not compiled, not run on hardware
//
#include <hip/hip_runtime.h>
#include <hip/hip_bf16.h>

// ---------------------------------------------------------------------------
// Types for CDNA5 WMMA (wave32, 16x16x32 bf16 -> f32)
// ---------------------------------------------------------------------------
typedef __bf16 bf16;
typedef __attribute__((ext_vector_type(16))) __bf16 v16bf;
typedef __attribute__((ext_vector_type(8)))  __bf16 v8bf;
typedef __attribute__((ext_vector_type(8)))  float  v8f;

#define BT_     4096        // B*T tokens
#define TLEN_   1024
#define NBATCH_ 4
#define HID_    512
#define NH_     8
#define NN_     64
#define FFN_    2048
#define LAYERS_ 6
#define LN_EPS_ 1e-5f
#define GN_EPS_ 0.04096f    // 1e-5 * 64^2

// ---------------------------------------------------------------------------
// CDNA5 async global->LDS copy (ASYNCcnt path), guarded for toolchain support
// ---------------------------------------------------------------------------
#if defined(__has_builtin)
#if __has_builtin(__builtin_amdgcn_global_load_async_to_lds_b128)
#define HAVE_ASYNC 1
#endif
#endif
#ifndef HAVE_ASYNC
#define HAVE_ASYNC 0
#endif

#if HAVE_ASYNC
typedef int i128v __attribute__((__vector_size__(16)));   // 16-byte int vector
__device__ __forceinline__ void async_b128(const bf16* g, bf16* l) {
  __builtin_amdgcn_global_load_async_to_lds_b128(
      (__attribute__((address_space(1))) i128v*)g,
      (__attribute__((address_space(3))) i128v*)l, 0, 0);
}
__device__ __forceinline__ void wait_async0() {
#if __has_builtin(__builtin_amdgcn_s_wait_asynccnt)
  __builtin_amdgcn_s_wait_asynccnt(0);
#else
  asm volatile("s_wait_asynccnt 0x0" ::: "memory");
#endif
}
#endif

// ---------------------------------------------------------------------------
// Block reductions
// ---------------------------------------------------------------------------
__device__ __forceinline__ float block_reduce64(float v, float* red, int tid) {
  red[tid] = v; __syncthreads();
#pragma unroll
  for (int s = 32; s > 0; s >>= 1) {
    if (tid < s) red[tid] += red[tid + s];
    __syncthreads();
  }
  float r = red[0]; __syncthreads();
  return r;
}

__device__ __forceinline__ float block_reduceN(float v, float* red, int tid, int n) {
  red[tid] = v; __syncthreads();
  for (int s = n >> 1; s > 0; s >>= 1) {
    if (tid < s) red[tid] += red[tid + s];
    __syncthreads();
  }
  float r = red[0]; __syncthreads();
  return r;
}

// ---------------------------------------------------------------------------
// WMMA GEMM: C[M,N] = act( A[M,K](f32) * Bt[N,K](bf16)^T + bias ) (+C if accum)
//   B operand pre-transposed: row n of Bt holds column n of B, K-contiguous.
//   Tile 128x128x32, 256 threads = 8 waves (4x2), wave tile 32x64 = 8 WMMA.
//   Double-buffered LDS, one barrier per K-step. B tile moved by async
//   global->LDS DMA when available; A tile converted f32->bf16 in registers.
//   ACT: 0=none 1=sigmoid 2=tanh 3=relu^2 4=rwkv decay exp(-exp(-sp(-z)-.5))
// ---------------------------------------------------------------------------
#define BM 128
#define BN 128
#define BK 32

__device__ __forceinline__ v8bf cvt8(float4 a, float4 b) {
  v8bf r;
  r[0] = (bf16)a.x; r[1] = (bf16)a.y; r[2] = (bf16)a.z; r[3] = (bf16)a.w;
  r[4] = (bf16)b.x; r[5] = (bf16)b.y; r[6] = (bf16)b.z; r[7] = (bf16)b.w;
  return r;
}

template <int ACT, bool ACCUM>
__global__ __launch_bounds__(256) void k_gemm(
    const float* __restrict__ A, const bf16* __restrict__ Bt,
    float* __restrict__ C, int M, int N, int K,
    int lda, int ldc, const float* __restrict__ bias)
{
  __shared__ bf16 sA[2][BM][BK];   // 2 x 8 KB
  __shared__ bf16 sB[2][BN][BK];   // 2 x 8 KB  ([n][k])

  const int tid  = threadIdx.x;
  const int m0   = blockIdx.y * BM;
  const int n0   = blockIdx.x * BN;
  const int wave = tid >> 5;
  const int lane = tid & 31;
  const int wm   = (wave & 3) * 32;   // wave row offset within tile
  const int wn   = (wave >> 2) * 64;  // wave col offset within tile
  const int lhalf = lane >> 4;        // 0 or 1
  const int lmod  = lane & 15;

  v8f acc[2][4];
#pragma unroll
  for (int i = 0; i < 2; i++)
#pragma unroll
    for (int j = 0; j < 4; j++) acc[i][j] = (v8f)0.0f;

  // staging assignments: each thread moves 16 A f32 and 16 Bt bf16
  const int arow = tid >> 1;           // 0..127
  const int acol = (tid & 1) * 16;     // 0 or 16
  const int brow = tid >> 1;           // 0..127
  const int bcol = (tid & 1) * 16;     // 0 or 16
  const bool binb = (n0 + brow) < N;

  const float* aptr = A + (size_t)(m0 + arow) * lda + acol;
  const bf16*  bptr = Bt + (size_t)(n0 + brow) * K + bcol;

  float4 fa0, fa1, fa2, fa3;
  uint4  ub0, ub1;
  ub0 = make_uint4(0, 0, 0, 0); ub1 = ub0;

  // ---- prologue: fetch tile k=0
  {
    const float4* sa = (const float4*)aptr;
    fa0 = sa[0]; fa1 = sa[1]; fa2 = sa[2]; fa3 = sa[3];
#if HAVE_ASYNC
    if (binb) {
      async_b128(bptr,     &sB[0][brow][bcol]);
      async_b128(bptr + 8, &sB[0][brow][bcol + 8]);
    }
#else
    if (binb) {
      const uint4* sb = (const uint4*)bptr;
      ub0 = sb[0]; ub1 = sb[1];
    }
#endif
  }
  *(v8bf*)&sA[0][arow][acol]     = cvt8(fa0, fa1);
  *(v8bf*)&sA[0][arow][acol + 8] = cvt8(fa2, fa3);
#if HAVE_ASYNC
  wait_async0();
#else
  *(uint4*)&sB[0][brow][bcol]     = ub0;
  *(uint4*)&sB[0][brow][bcol + 8] = ub1;
#endif
  __syncthreads();

  int p = 0;
  for (int k0 = 0; k0 < K; k0 += BK) {
    const bool hasNext = (k0 + BK) < K;
    if (hasNext) {
      const float4* sa = (const float4*)(aptr + k0 + BK);
      fa0 = sa[0]; fa1 = sa[1]; fa2 = sa[2]; fa3 = sa[3];
#if HAVE_ASYNC
      if (binb) {
        async_b128(bptr + k0 + BK,     &sB[p ^ 1][brow][bcol]);
        async_b128(bptr + k0 + BK + 8, &sB[p ^ 1][brow][bcol + 8]);
      }
#else
      if (binb) {
        const uint4* sb = (const uint4*)(bptr + k0 + BK);
        ub0 = sb[0]; ub1 = sb[1];
      } else {
        ub0 = make_uint4(0, 0, 0, 0); ub1 = ub0;
      }
#endif
      if (k0 + 2 * BK < K)
        __builtin_prefetch(aptr + k0 + 2 * BK, 0, 1);
    }

    // ---- fragments per ISA 16-bit layouts
    v16bf afrag[2], bfrag[4];
#pragma unroll
    for (int i = 0; i < 2; i++) {
      int m = wm + i * 16 + lmod;
      v8bf lo = *(const v8bf*)&sA[p][m][8 * lhalf];        // K = 8*half .. +7
      v8bf hi = *(const v8bf*)&sA[p][m][16 + 8 * lhalf];   // K = 16+8*half .. +7
#pragma unroll
      for (int e = 0; e < 8; e++) { afrag[i][e] = lo[e]; afrag[i][e + 8] = hi[e]; }
    }
#pragma unroll
    for (int j = 0; j < 4; j++) {
      int n = wn + j * 16 + lmod;
      v8bf lo = *(const v8bf*)&sB[p][n][8 * lhalf];        // K = 8*half .. +7
      v8bf hi = *(const v8bf*)&sB[p][n][16 + 8 * lhalf];   // K = 16+8*half .. +7
#pragma unroll
      for (int e = 0; e < 8; e++) { bfrag[j][e] = lo[e]; bfrag[j][e + 8] = hi[e]; }
    }
#pragma unroll
    for (int i = 0; i < 2; i++)
#pragma unroll
      for (int j = 0; j < 4; j++)
        acc[i][j] = __builtin_amdgcn_wmma_f32_16x16x32_bf16(
            false, afrag[i], false, bfrag[j], (short)0, acc[i][j], false, false);

    if (hasNext) {
      const int q = p ^ 1;
      *(v8bf*)&sA[q][arow][acol]      = cvt8(fa0, fa1);
      *(v8bf*)&sA[q][arow][acol + 8]  = cvt8(fa2, fa3);
#if HAVE_ASYNC
      wait_async0();
#else
      *(uint4*)&sB[q][brow][bcol]     = ub0;
      *(uint4*)&sB[q][brow][bcol + 8] = ub1;
#endif
      __syncthreads();
      p = q;
    }
  }

  // ---- epilogue + store (C/D layout: VGPR v holds M=v / v+8, N = lane&15)
#pragma unroll
  for (int i = 0; i < 2; i++) {
#pragma unroll
    for (int j = 0; j < 4; j++) {
      int col = n0 + wn + j * 16 + lmod;
      if (col < N) {
        float bv = bias ? bias[col] : 0.0f;
#pragma unroll
        for (int v = 0; v < 8; v++) {
          int row = m0 + wm + i * 16 + lhalf * 8 + v;
          float val = acc[i][j][v] + bv;
          if (ACT == 1) {
            val = 1.0f / (1.0f + __expf(-val));
          } else if (ACT == 2) {
            val = tanhf(val);
          } else if (ACT == 3) {
            val = fmaxf(val, 0.0f); val = val * val;
          } else if (ACT == 4) {
            float w = -__logf(1.0f + __expf(-val)) - 0.5f;  // -softplus(-z)-0.5
            val = __expf(-__expf(w));                        // decay
          }
          float* pc = C + (size_t)row * ldc + col;
          if (ACCUM) val += *pc;
          *pc = val;
        }
      }
    }
  }
}

// ---------------------------------------------------------------------------
// Weight preparation kernels
// ---------------------------------------------------------------------------
__global__ void k_f32_bf16(const float* __restrict__ src, bf16* __restrict__ dst, size_t n) {
  size_t i = (size_t)blockIdx.x * 256 + threadIdx.x;
  if (i < n) dst[i] = (bf16)src[i];
}

// src [K,N] f32 row-major -> dst [N,K] bf16 row-major (B^T), LDS-tiled 32x32
__global__ __launch_bounds__(256) void k_convT(const float* __restrict__ src,
                                               bf16* __restrict__ dst, int K, int N) {
  __shared__ bf16 tile[32][33];
  const int kb = blockIdx.x * 32, nb = blockIdx.y * 32;
  const int tx = threadIdx.x & 31, ty = threadIdx.x >> 5;   // 32 x 8
#pragma unroll
  for (int r = 0; r < 32; r += 8) {
    int k = kb + ty + r, n = nb + tx;
    tile[ty + r][tx] = (k < K && n < N) ? (bf16)src[(size_t)k * N + n] : (bf16)0.0f;
  }
  __syncthreads();
#pragma unroll
  for (int r = 0; r < 32; r += 8) {
    int n = nb + ty + r, k = kb + tx;
    if (n < N && k < K) dst[(size_t)n * K + k] = tile[tx][ty + r];
  }
}

// Wm^T rows: dst[o][c] = mask_g[o] * v[o][c]/||v[o]|| (already [N=o][K=c])
__global__ __launch_bounds__(128) void k_masknorm(
    const float* __restrict__ v, const float* __restrict__ mg, bf16* __restrict__ dst) {
  __shared__ float red[128];
  int o = blockIdx.x, tid = threadIdx.x;
  const float* row = v + (size_t)o * 512;
  float ss = 0.0f;
  for (int c = tid; c < 512; c += 128) { float x = row[c]; ss += x * x; }
  ss = block_reduceN(ss, red, tid, 128);
  float scale = mg[o] / sqrtf(ss);
  for (int c = tid; c < 512; c += 128) dst[(size_t)o * 512 + c] = (bf16)(row[c] * scale);
}

// ---------------------------------------------------------------------------
// Elementwise / normalization kernels
// ---------------------------------------------------------------------------
__global__ __launch_bounds__(256) void k_layernorm(
    const float* __restrict__ x, const float* __restrict__ g, const float* __restrict__ b,
    float* __restrict__ out) {
  __shared__ float red[256];
  int row = blockIdx.x, tid = threadIdx.x;
  const float* xr = x + (size_t)row * 512;
  float v0 = xr[tid], v1 = xr[tid + 256];
  float mu = block_reduceN(v0 + v1, red, tid, 256) * (1.0f / 512.0f);
  float d0 = v0 - mu, d1 = v1 - mu;
  float var = block_reduceN(d0 * d0 + d1 * d1, red, tid, 256) * (1.0f / 512.0f);
  float inv = rsqrtf(var + LN_EPS_);
  float* orow = out + (size_t)row * 512;
  orow[tid]       = d0 * inv * g[tid]       + b[tid];
  orow[tid + 256] = d1 * inv * g[tid + 256] + b[tid + 256];
}

__global__ void k_shift_mix6(
    const float* __restrict__ x,
    const float* __restrict__ cr, const float* __restrict__ cw, const float* __restrict__ ck,
    const float* __restrict__ cv, const float* __restrict__ ca, const float* __restrict__ cg,
    float* __restrict__ o_r, float* __restrict__ o_w, float* __restrict__ o_k,
    float* __restrict__ o_v, float* __restrict__ o_a, float* __restrict__ o_g) {
  int idx = blockIdx.x * 256 + threadIdx.x;
  if (idx >= BT_ * 512) return;
  int c = idx & 511;
  int token = idx >> 9;
  int tloc = token & (TLEN_ - 1);
  float xv = x[idx];
  float prev = tloc ? x[idx - 512] : 0.0f;
  float xx = prev - xv;
  o_r[idx] = xv + xx * cr[c];
  o_w[idx] = xv + xx * cw[c];
  o_k[idx] = xv + xx * ck[c];
  o_v[idx] = xv + xx * cv[c];
  o_a[idx] = xv + xx * ca[c];
  o_g[idx] = xv + xx * cg[c];
}

__global__ void k_shift_mix1(const float* __restrict__ x, const float* __restrict__ ck,
                             float* __restrict__ out) {
  int idx = blockIdx.x * 256 + threadIdx.x;
  if (idx >= BT_ * 512) return;
  int c = idx & 511;
  int token = idx >> 9;
  int tloc = token & (TLEN_ - 1);
  float xv = x[idx];
  float prev = tloc ? x[idx - 512] : 0.0f;
  out[idx] = xv + (prev - xv) * ck[c];
}

__global__ void k_vmix(float* __restrict__ v, const float* __restrict__ vf,
                       const float* __restrict__ sig, int n) {
  int idx = blockIdx.x * 256 + threadIdx.x;
  if (idx >= n) return;
  float vv = v[idx];
  v[idx] = vv + (vf[idx] - vv) * sig[idx];
}

// kk normalize + a/b/kmod prep, one 64-thread block per (token, head)
__global__ __launch_bounds__(64) void k_kkprep(
    const float* __restrict__ k, const float* __restrict__ a,
    const float* __restrict__ k_k, const float* __restrict__ k_a,
    float* __restrict__ aa, float* __restrict__ bb, float* __restrict__ kmod) {
  __shared__ float red[64];
  int bh = blockIdx.x;            // token*8 + head
  int head = bh & 7, i = threadIdx.x;
  size_t off = (size_t)bh * 64 + i;
  int c = head * 64 + i;
  float kv = k[off], av = a[off];
  float kkv = kv * k_k[c];
  float ss = block_reduce64(kkv * kkv, red, i);
  float nrm = fmaxf(sqrtf(ss), 1e-12f);
  float kkn = kkv / nrm;
  aa[off]   = -kkn;
  bb[off]   = kkn * av;
  kmod[off] = kv * (1.0f + (av - 1.0f) * k_a[c]);
}

// RWKV-v7 matrix-state scan. One block per (batch, head), 64 threads,
// thread i owns state row S[i][0..63] in VGPRs.
__global__ __launch_bounds__(64) void k_wkv(
    const float* __restrict__ r, const float* __restrict__ d,
    const float* __restrict__ k, const float* __restrict__ v,
    const float* __restrict__ a, const float* __restrict__ b,
    float* __restrict__ y) {
  int bh = blockIdx.x;                   // batch*8 + head
  int batch = bh >> 3, head = bh & 7;
  int i = threadIdx.x;
  __shared__ float rs[64], ds[64], ks[64], as[64], bs[64];
  float S[64];
#pragma unroll
  for (int j = 0; j < 64; j++) S[j] = 0.0f;
  size_t base = ((size_t)batch * TLEN_) * 512 + head * 64;
  for (int t = 0; t < TLEN_; t++) {
    size_t off = base + (size_t)t * 512 + i;
    rs[i] = r[off]; ds[i] = d[off]; ks[i] = k[off]; as[i] = a[off]; bs[i] = b[off];
    float vi = v[off];
    __syncthreads();
    float sa = 0.0f;
#pragma unroll
    for (int j = 0; j < 64; j++) sa += S[j] * as[j];
    float yi = 0.0f;
#pragma unroll
    for (int j = 0; j < 64; j++) {
      S[j] = S[j] * ds[j] + sa * bs[j] + vi * ks[j];
      yi += S[j] * rs[j];
    }
    y[off] = yi;
    __syncthreads();
  }
}

// groupnorm + bonus + gate, one 64-thread block per (token, head)
__global__ __launch_bounds__(64) void k_gn_bonus_gate(
    const float* __restrict__ y, const float* __restrict__ r, const float* __restrict__ k,
    const float* __restrict__ v, const float* __restrict__ g, const float* __restrict__ r_k,
    const float* __restrict__ lnx_g, const float* __restrict__ lnx_b, float* __restrict__ out) {
  __shared__ float red[64];
  int bh = blockIdx.x;
  int head = bh & 7, i = threadIdx.x;
  size_t off = (size_t)bh * 64 + i;
  float yv = y[off];
  float mu = block_reduce64(yv, red, i) * (1.0f / 64.0f);
  float dd = yv - mu;
  float var = block_reduce64(dd * dd, red, i) * (1.0f / 64.0f);
  float gn = dd * rsqrtf(var + GN_EPS_);
  float dot = block_reduce64(r[off] * k[off] * r_k[head * 64 + i], red, i);
  int c = head * 64 + i;
  out[off] = (gn * lnx_g[c] + lnx_b[c] + dot * v[off]) * g[off];
}

__global__ void k_half(const float* __restrict__ a, const float* __restrict__ b,
                       float* __restrict__ o, int n) {
  int idx = blockIdx.x * 256 + threadIdx.x;
  if (idx < n) o[idx] = 0.5f * (a[idx] + b[idx]);
}

__global__ void k_flip(const float* __restrict__ in, float* __restrict__ out) {
  int idx = blockIdx.x * 256 + threadIdx.x;
  if (idx >= BT_ * 512) return;
  int c = idx & 511;
  int token = idx >> 9;
  int b = token >> 10, t = token & 1023;
  out[((((size_t)b << 10) + (1023 - t)) << 9) + c] = in[idx];
}

// snake(content, ha) * sigmoid(gate) -> snake(., act_alpha)
__global__ void k_head_out(const float* __restrict__ content, const float* __restrict__ gate,
                           const float* __restrict__ head_alpha, int spk,
                           const float* __restrict__ act_alpha, float* __restrict__ out, int n) {
  int idx = blockIdx.x * 256 + threadIdx.x;
  if (idx >= n) return;
  float a1 = head_alpha[spk] + 1e-9f;
  float z = content[idx];
  float s1 = __sinf(a1 * z);
  float c = z + s1 * s1 / a1;
  float gt = 1.0f / (1.0f + __expf(-gate[idx]));
  float xx = c * gt;
  float a2 = act_alpha[0] + 1e-9f;
  float s2 = __sinf(a2 * xx);
  out[idx] = xx + s2 * s2 / a2;
}

// ---------------------------------------------------------------------------
// Host orchestration
// ---------------------------------------------------------------------------
enum { iW_k = 0, iW_key, iW_o, iW_r, iW_v, iW_val, ia0, ia1, ia2, icx_k, ig1, ig2,
       ik_a, ik_k, iln1_b, iln1_g, iln2_b, iln2_g, ilnx_b, ilnx_g, ir_k,
       iv0, iv1, iv2, iw0, iw1, iw2, ix_a, ix_g, ix_k, ix_r, ix_v, ix_w, NLP };

struct LayerP { const float* p[NLP]; };
struct LayerW { bf16 *Wr, *Wk, *Wv, *Wo, *w1, *w2, *a1, *a2, *v1, *v2, *g1, *g2, *Wkey, *Wval; };

struct Scratch {
  float *x0, *xf, *xb, *tflip, *tln;
  float *txr, *txw, *txk, *txv, *txa, *txg;
  float *tr, *tk, *tv, *tdecay, *ta, *tg, *tvsig, *taa, *tbb, *tkmod, *ty, *tyg, *vfirst;
  float *tlr, *tffn, *tkx, *th, *tup, *tmask, *tcontent, *tgate;
};

struct Arena {
  char* base; size_t off;
  void* get(size_t bytes) {
    off = (off + 255) & ~(size_t)255;
    void* p = base + off;
    off += bytes;
    return p;
  }
};

// act: 0=none 1=sigmoid 2=tanh 3=relu^2 4=decay ; accum adds into C
static inline void gemm(hipStream_t s, const float* A, const bf16* Bt, float* C,
                        int M, int N, int K, int lda, int ldc,
                        const float* bias, int act, int accum) {
  dim3 g((N + BN - 1) / BN, (M + BM - 1) / BM);
  if (accum) {
    k_gemm<0, true><<<g, 256, 0, s>>>(A, Bt, C, M, N, K, lda, ldc, bias);
    return;
  }
  switch (act) {
    case 1:  k_gemm<1, false><<<g, 256, 0, s>>>(A, Bt, C, M, N, K, lda, ldc, bias); break;
    case 2:  k_gemm<2, false><<<g, 256, 0, s>>>(A, Bt, C, M, N, K, lda, ldc, bias); break;
    case 3:  k_gemm<3, false><<<g, 256, 0, s>>>(A, Bt, C, M, N, K, lda, ldc, bias); break;
    case 4:  k_gemm<4, false><<<g, 256, 0, s>>>(A, Bt, C, M, N, K, lda, ldc, bias); break;
    default: k_gemm<0, false><<<g, 256, 0, s>>>(A, Bt, C, M, N, K, lda, ldc, bias); break;
  }
}

static void run_core(hipStream_t s, const LayerP* lp, const LayerW* lw,
                     float* x, const Scratch& W) {
  const int M = BT_;
  const int NEL = M * 512;
  const int GE = (NEL + 255) / 256;
  for (int L = 0; L < LAYERS_; ++L) {
    const LayerP& p = lp[L];
    const LayerW& w = lw[L];
    // ---- time mix
    k_layernorm<<<M, 256, 0, s>>>(x, p.p[iln1_g], p.p[iln1_b], W.tln);
    k_shift_mix6<<<GE, 256, 0, s>>>(W.tln, p.p[ix_r], p.p[ix_w], p.p[ix_k],
                                    p.p[ix_v], p.p[ix_a], p.p[ix_g],
                                    W.txr, W.txw, W.txk, W.txv, W.txa, W.txg);
    gemm(s, W.txr, w.Wr, W.tr, M, 512, 512, 512, 512, nullptr, 0, 0);
    gemm(s, W.txk, w.Wk, W.tk, M, 512, 512, 512, 512, nullptr, 0, 0);
    gemm(s, W.txv, w.Wv, W.tv, M, 512, 512, 512, 512, nullptr, 0, 0);
    gemm(s, W.txw, w.w1, W.tlr, M, 64, 512, 512, 64, nullptr, 2, 0);           // tanh
    gemm(s, W.tlr, w.w2, W.tdecay, M, 512, 64, 64, 512, p.p[iw0], 4, 0);       // decay
    gemm(s, W.txa, w.a1, W.tlr, M, 64, 512, 512, 64, nullptr, 0, 0);
    gemm(s, W.tlr, w.a2, W.ta, M, 512, 64, 64, 512, p.p[ia0], 1, 0);           // sigmoid
    gemm(s, W.txg, w.g1, W.tlr, M, 128, 512, 512, 128, nullptr, 1, 0);         // sigmoid
    gemm(s, W.tlr, w.g2, W.tg, M, 512, 128, 128, 512, nullptr, 0, 0);
    if (L == 0) {
      (void)hipMemcpyAsync(W.vfirst, W.tv, (size_t)NEL * sizeof(float),
                           hipMemcpyDeviceToDevice, s);
    } else {
      gemm(s, W.txv, w.v1, W.tlr, M, 32, 512, 512, 32, nullptr, 0, 0);
      gemm(s, W.tlr, w.v2, W.tvsig, M, 512, 32, 32, 512, p.p[iv0], 1, 0);      // sigmoid
      k_vmix<<<GE, 256, 0, s>>>(W.tv, W.vfirst, W.tvsig, NEL);
    }
    k_kkprep<<<M * NH_, 64, 0, s>>>(W.tk, W.ta, p.p[ik_k], p.p[ik_a],
                                    W.taa, W.tbb, W.tkmod);
    k_wkv<<<NBATCH_ * NH_, 64, 0, s>>>(W.tr, W.tdecay, W.tkmod, W.tv, W.taa, W.tbb, W.ty);
    k_gn_bonus_gate<<<M * NH_, 64, 0, s>>>(W.ty, W.tr, W.tkmod, W.tv, W.tg,
                                           p.p[ir_k], p.p[ilnx_g], p.p[ilnx_b], W.tyg);
    gemm(s, W.tyg, w.Wo, x, M, 512, 512, 512, 512, nullptr, 0, 1);             // x += h
    // ---- channel mix
    k_layernorm<<<M, 256, 0, s>>>(x, p.p[iln2_g], p.p[iln2_b], W.tln);
    k_shift_mix1<<<GE, 256, 0, s>>>(W.tln, p.p[icx_k], W.tkx);
    gemm(s, W.tkx, w.Wkey, W.tffn, M, 2048, 512, 512, 2048, nullptr, 3, 0);    // relu^2
    gemm(s, W.tffn, w.Wval, x, M, 512, 2048, 2048, 512, nullptr, 0, 1);        // x += ffn
  }
}

extern "C" void kernel_launch(void* const* d_in, const int* in_sizes, int n_in,
                              void* d_out, int out_size, void* d_ws, size_t ws_size,
                              hipStream_t stream) {
  (void)in_sizes; (void)n_in; (void)out_size; (void)ws_size;

  // ---- input mapping: z_mix first, then params flattened with sorted keys
  // params top-level order: act_alpha, bwd[6x33], down_b, down_w, fwd[6x33],
  //                         gate_w, head_alpha, head_w, mask_g, mask_v, up_b, up_w
  const float* z_mix = (const float*)d_in[0];
  const int IDX_ACT_ALPHA = 1;
  const int IDX_BWD0 = 2;
  const int IDX_DOWN_B = 2 + 6 * NLP;          // 200
  const int IDX_DOWN_W = IDX_DOWN_B + 1;       // 201
  const int IDX_FWD0   = IDX_DOWN_W + 1;       // 202
  const int IDX_GATE_W = IDX_FWD0 + 6 * NLP;   // 400
  const int IDX_HEAD_ALPHA = IDX_GATE_W + 1;
  const int IDX_HEAD_W = IDX_GATE_W + 2;
  const int IDX_MASK_G = IDX_GATE_W + 3;
  const int IDX_MASK_V = IDX_GATE_W + 4;
  const int IDX_UP_B   = IDX_GATE_W + 5;
  const int IDX_UP_W   = IDX_GATE_W + 6;

  LayerP fwd[LAYERS_], bwd[LAYERS_];
  for (int L = 0; L < LAYERS_; L++)
    for (int q = 0; q < NLP; q++) {
      bwd[L].p[q] = (const float*)d_in[IDX_BWD0 + L * NLP + q];
      fwd[L].p[q] = (const float*)d_in[IDX_FWD0 + L * NLP + q];
    }
  const float* act_alpha  = (const float*)d_in[IDX_ACT_ALPHA];
  const float* down_b     = (const float*)d_in[IDX_DOWN_B];
  const float* down_w     = (const float*)d_in[IDX_DOWN_W];
  const float* gate_w     = (const float*)d_in[IDX_GATE_W];
  const float* head_alpha = (const float*)d_in[IDX_HEAD_ALPHA];
  const float* head_w     = (const float*)d_in[IDX_HEAD_W];
  const float* mask_g     = (const float*)d_in[IDX_MASK_G];
  const float* mask_v     = (const float*)d_in[IDX_MASK_V];
  const float* up_b       = (const float*)d_in[IDX_UP_B];
  const float* up_w       = (const float*)d_in[IDX_UP_W];

  Arena A{(char*)d_ws, 0};

  // ---- prepare weights as bf16 B^T ([N,K] row-major, K-contiguous)
  auto cvtT = [&](const float* src, int K, int N) -> bf16* {
    bf16* d = (bf16*)A.get((size_t)K * N * sizeof(bf16));
    dim3 g((K + 31) / 32, (N + 31) / 32);
    k_convT<<<g, 256, 0, stream>>>(src, d, K, N);
    return d;
  };
  auto cvtFlat = [&](const float* src, size_t n) -> bf16* {  // already [N,K]
    bf16* d = (bf16*)A.get(n * sizeof(bf16));
    k_f32_bf16<<<(int)((n + 255) / 256), 256, 0, stream>>>(src, d, n);
    return d;
  };
  LayerW lwf[LAYERS_], lwb[LAYERS_];
  for (int dir = 0; dir < 2; dir++) {
    LayerP* lp = dir ? bwd : fwd;
    LayerW* lw = dir ? lwb : lwf;
    for (int L = 0; L < LAYERS_; L++) {
      lw[L].Wr   = cvtT(lp[L].p[iW_r],   512, 512);
      lw[L].Wk   = cvtT(lp[L].p[iW_k],   512, 512);
      lw[L].Wv   = cvtT(lp[L].p[iW_v],   512, 512);
      lw[L].Wo   = cvtT(lp[L].p[iW_o],   512, 512);
      lw[L].w1   = cvtT(lp[L].p[iw1],    512, 64);
      lw[L].w2   = cvtT(lp[L].p[iw2],    64, 512);
      lw[L].a1   = cvtT(lp[L].p[ia1],    512, 64);
      lw[L].a2   = cvtT(lp[L].p[ia2],    64, 512);
      lw[L].v1   = cvtT(lp[L].p[iv1],    512, 32);
      lw[L].v2   = cvtT(lp[L].p[iv2],    32, 512);
      lw[L].g1   = cvtT(lp[L].p[ig1],    512, 128);
      lw[L].g2   = cvtT(lp[L].p[ig2],    128, 512);
      lw[L].Wkey = cvtT(lp[L].p[iW_key], 512, 2048);
      lw[L].Wval = cvtT(lp[L].p[iW_val], 2048, 512);
    }
  }
  bf16* downw = cvtT(down_w, 512, 512);
  bf16* upw   = cvtT(up_w,   512, 512);
  bf16* maskB = (bf16*)A.get((size_t)1024 * 512 * sizeof(bf16));   // [N=1024][K=512]
  k_masknorm<<<1024, 128, 0, stream>>>(mask_v, mask_g, maskB);
  bf16 *headB[2], *gateB[2];
  for (int i = 0; i < 2; i++) {
    headB[i] = cvtFlat(head_w + (size_t)i * 512 * 512, (size_t)512 * 512);  // [O,C]=[N,K]
    gateB[i] = cvtFlat(gate_w + (size_t)i * 512 * 512, (size_t)512 * 512);
  }

  // ---- activation scratch
  Scratch S;
  auto fb = [&](size_t n) -> float* { return (float*)A.get(n * sizeof(float)); };
  const size_t NE = (size_t)BT_ * 512;
  S.x0 = fb(NE); S.xf = fb(NE); S.xb = fb(NE); S.tflip = fb(NE); S.tln = fb(NE);
  S.txr = fb(NE); S.txw = fb(NE); S.txk = fb(NE); S.txv = fb(NE); S.txa = fb(NE); S.txg = fb(NE);
  S.tr = fb(NE); S.tk = fb(NE); S.tv = fb(NE); S.tdecay = fb(NE); S.ta = fb(NE); S.tg = fb(NE);
  S.tvsig = fb(NE); S.taa = fb(NE); S.tbb = fb(NE); S.tkmod = fb(NE); S.ty = fb(NE);
  S.tyg = fb(NE); S.vfirst = fb(NE);
  S.tlr = fb((size_t)BT_ * 128);
  S.tffn = fb((size_t)BT_ * 2048);
  S.tkx = fb(NE); S.th = fb(NE); S.tup = fb(NE);
  S.tmask = fb((size_t)BT_ * 1024);
  S.tcontent = fb(NE); S.tgate = fb(NE);

  const int GE = (int)((NE + 255) / 256);

  // ---- pipeline
  gemm(stream, z_mix, downw, S.x0, BT_, 512, 512, 512, 512, down_b, 0, 0);
  (void)hipMemcpyAsync(S.xf, S.x0, NE * sizeof(float), hipMemcpyDeviceToDevice, stream);
  run_core(stream, fwd, lwf, S.xf, S);
  k_flip<<<GE, 256, 0, stream>>>(S.x0, S.xb);
  run_core(stream, bwd, lwb, S.xb, S);
  k_flip<<<GE, 256, 0, stream>>>(S.xb, S.tflip);
  k_half<<<GE, 256, 0, stream>>>(S.xf, S.tflip, S.th, (int)NE);

  gemm(stream, S.th, upw, S.tup, BT_, 512, 512, 512, 512, up_b, 0, 0);
  gemm(stream, S.tup, maskB, S.tmask, BT_, 1024, 512, 512, 1024, nullptr, 0, 0);

  float* out = (float*)d_out;
  for (int i = 0; i < 2; i++) {
    const float* sA = S.tmask + (size_t)i * 512;   // strided rows, lda=1024
    gemm(stream, sA, headB[i], S.tcontent, BT_, 512, 512, 1024, 512, nullptr, 0, 0);
    gemm(stream, sA, gateB[i], S.tgate,    BT_, 512, 512, 1024, 512, nullptr, 0, 0);
    k_head_out<<<GE, 256, 0, stream>>>(S.tcontent, S.tgate, head_alpha, i, act_alpha,
                                       out + (size_t)i * NE, (int)NE);
  }
}
